// TFJSPModel_hgnn_21414706937970
// MI455X (gfx1250) — compile-verified, matrix operand order checked by reading, not run on hardware
//
#include <hip/hip_runtime.h>
#include <hip/hip_bf16.h>

typedef __attribute__((ext_vector_type(16))) _Float16 v16h;
typedef __attribute__((ext_vector_type(8)))  float    v8f;
typedef __attribute__((ext_vector_type(4)))  unsigned int v4u;
typedef __attribute__((ext_vector_type(8)))  int      v8i;
typedef __attribute__((ext_vector_type(4)))  int      v4i;

static constexpr int CB = 8;      // batch
static constexpr int CO = 2000;   // opes
static constexpr int CM = 30;     // machines
static constexpr int CJ = 200;    // jobs

__device__ inline float lrelu(float x) { return x > 0.f ? x : 0.2f * x; }

__device__ inline float blk_sum(float v, float* red) {
    int t = threadIdx.x;
    red[t] = v; __syncthreads();
    for (int s = blockDim.x >> 1; s > 0; s >>= 1) {
        if (t < s) red[t] += red[t + s];
        __syncthreads();
    }
    float r = red[0]; __syncthreads();
    return r;
}
__device__ inline float blk_max(float v, float* red) {
    int t = threadIdx.x;
    red[t] = v; __syncthreads();
    for (int s = blockDim.x >> 1; s > 0; s >>= 1) {
        if (t < s) red[t] = fmaxf(red[t], red[t + s]);
        __syncthreads();
    }
    float r = red[0]; __syncthreads();
    return r;
}

// ---------------- normalization over node axis (ddof=1) ----------------
__global__ void norm_nodes(const float* __restrict__ x, float* __restrict__ out,
                           int Nn, int D) {
    int b = blockIdx.x / D, f = blockIdx.x % D;
    __shared__ float rs[256];
    __shared__ float smu, srs;
    float s = 0.f, q = 0.f;
    for (int i = threadIdx.x; i < Nn; i += blockDim.x) {
        float v = x[((size_t)b * Nn + i) * D + f];
        s += v; q += v * v;
    }
    float ts = blk_sum(s, rs);
    float tq = blk_sum(q, rs);
    if (threadIdx.x == 0) {
        float mu = ts / Nn;
        float var = (tq - Nn * mu * mu) / (Nn - 1);
        smu = mu;
        srs = 1.f / (sqrtf(fmaxf(var, 0.f)) + 1e-5f);
    }
    __syncthreads();
    for (int i = threadIdx.x; i < Nn; i += blockDim.x) {
        float v = x[((size_t)b * Nn + i) * D + f];
        out[((size_t)b * Nn + i) * 8 + f] = (v - smu) * srs;
    }
}

// ---------------- zero the padding columns [dfrom,8) of a stride-8 node buffer -------------
__global__ void zero_pad_cols(float* __restrict__ X, int total, int dfrom) {
    int i = blockIdx.x * blockDim.x + threadIdx.x;
    if (i >= total) return;
    for (int k = dfrom; k < 8; ++k) X[(size_t)i * 8 + k] = 0.f;
}

// ---------------- proc_time global mean/std (ddof=1) ----------------
__global__ void proc_stats(const float* __restrict__ p, float* __restrict__ stats, int n) {
    __shared__ float rs[256];
    float s = 0.f, q = 0.f;
    for (int i = threadIdx.x; i < n; i += blockDim.x) {
        float v = p[i]; s += v; q += v * v;
    }
    float ts = blk_sum(s, rs);
    float tq = blk_sum(q, rs);
    if (threadIdx.x == 0) {
        float mu = ts / n;
        float var = (tq - n * mu * mu) / (n - 1);
        stats[0] = mu;
        stats[1] = 1.f / (sqrtf(fmaxf(var, 0.f)) + 1e-5f);
    }
}

// ---------------- GAT: fs = x_ope @ Wsrc, el = <fs, al> ----------------
__global__ void gat_src(const float* __restrict__ x, const float* __restrict__ Wsrc,
                        const float* __restrict__ al, float* __restrict__ fs,
                        float* __restrict__ el, int total, int din) {
    int i = blockIdx.x * blockDim.x + threadIdx.x;
    if (i >= total) return;
    float f[8];
    for (int j = 0; j < 8; ++j) {
        float s = 0.f;
        for (int k = 0; k < din; ++k) s += x[(size_t)i * 8 + k] * Wsrc[k * 8 + j];
        f[j] = s;
        fs[(size_t)i * 8 + j] = s;
    }
    float e = 0.f;
    for (int j = 0; j < 8; ++j) e += f[j] * al[j];
    el[i] = e;
}

// ---------------- GAT attention + aggregation, one block per (b,m) ----------------
__global__ void gat_attn(const float* __restrict__ xma, const float* __restrict__ Wdst,
                         const float* __restrict__ Wedge, const float* __restrict__ ae,
                         const float* __restrict__ ar, const float* __restrict__ fs,
                         const float* __restrict__ el, const float* __restrict__ proc,
                         const int* __restrict__ adj, const float* __restrict__ stats,
                         float* __restrict__ hmas, int dma) {
    int b = blockIdx.x / CM, m = blockIdx.x % CM;
    __shared__ float red[256];
    __shared__ float sacc[8];
    float fd[8];
    for (int j = 0; j < 8; ++j) {
        float s = 0.f;
        for (int k = 0; k < dma; ++k) s += xma[((size_t)b * CM + m) * 8 + k] * Wdst[k * 8 + j];
        fd[j] = s;
    }
    float er = 0.f;
    for (int j = 0; j < 8; ++j) er += fd[j] * ar[j];
    float wsum = 0.f;
    for (int j = 0; j < 8; ++j) wsum += Wedge[j] * ae[j];
    float pm = stats[0], pinv = stats[1];
    float ekk = lrelu(2.f * er);

    float lmax = ekk;
    for (int o = threadIdx.x; o < CO; o += blockDim.x) {
        if (adj[((size_t)b * CO + o) * CM + m] == 1) {
            float pn = (proc[((size_t)b * CO + o) * CM + m] - pm) * pinv;
            float e = lrelu(el[(size_t)b * CO + o] + pn * wsum + er);
            lmax = fmaxf(lmax, e);
        }
    }
    float gmax = blk_max(lmax, red);

    float lsum = 0.f;
    float acc[8] = {0.f, 0.f, 0.f, 0.f, 0.f, 0.f, 0.f, 0.f};
    for (int o = threadIdx.x; o < CO; o += blockDim.x) {
        if (adj[((size_t)b * CO + o) * CM + m] == 1) {
            float pn = (proc[((size_t)b * CO + o) * CM + m] - pm) * pinv;
            float e = lrelu(el[(size_t)b * CO + o] + pn * wsum + er);
            float w = expf(e - gmax);
            lsum += w;
            for (int j = 0; j < 8; ++j)
                acc[j] += w * (pn * Wedge[j] + fs[((size_t)b * CO + o) * 8 + j]);
        }
    }
    float wkk = expf(ekk - gmax);
    float gsum = blk_sum(lsum, red) + wkk;
    for (int j = 0; j < 8; ++j) {
        float aj = blk_sum(acc[j], red);
        if (threadIdx.x == 0) sacc[j] = aj;
    }
    __syncthreads();
    if (threadIdx.x < 8) {
        int j = threadIdx.x;
        float bv = sacc[j] / gsum;
        float akk = wkk / gsum;
        float z = bv + fd[j] * akk;
        hmas[((size_t)b * CM + m) * 8 + j] = 1.f / (1.f + expf(-z));
    }
}

// ---------------- agg0 = adj_ma @ h_mas ----------------
__global__ void agg_ma(const int* __restrict__ adj, const float* __restrict__ hmas,
                       float* __restrict__ out) {
    int i = blockIdx.x * blockDim.x + threadIdx.x;
    if (i >= CB * CO) return;
    int b = i / CO;
    float acc[8] = {0.f, 0.f, 0.f, 0.f, 0.f, 0.f, 0.f, 0.f};
    for (int m = 0; m < CM; ++m)
        if (adj[(size_t)i * CM + m] == 1)
            for (int j = 0; j < 8; ++j) acc[j] += hmas[((size_t)b * CM + m) * 8 + j];
    for (int j = 0; j < 8; ++j) out[(size_t)i * 8 + j] = acc[j];
}

// -------- device helper: issue one TDM tile load (16 x 32 int32) into LDS ---------------
__device__ inline void tdm_load_tile(const int* gsrc, int* ldst, unsigned td0, unsigned td1,
                                     unsigned tile0) {
#if __has_builtin(__builtin_amdgcn_tensor_load_to_lds) && __has_builtin(__builtin_amdgcn_s_wait_tensorcnt)
    // Tensor DMA descriptor (D#) per CDNA5 ISA:
    // group0: count=1 | lds_addr | global_addr(tile start) | type=2
    // group1: data_size=4B, tensor_dim0/1 = extent remaining from tile start
    //         (OOB reads return zero => implicit zero padding), tile 32x16,
    //         tensor_dim0_stride = CO
    unsigned long long ga = (unsigned long long)(const void*)gsrc;
    unsigned lds = (unsigned)(unsigned long long)(void*)ldst;
    v4u g0;
    g0[0] = 1u;
    g0[1] = (unsigned)__builtin_amdgcn_readfirstlane((int)lds);
    g0[2] = (unsigned)__builtin_amdgcn_readfirstlane((int)(unsigned)(ga & 0xffffffffull));
    g0[3] = (unsigned)__builtin_amdgcn_readfirstlane(
                (int)(((unsigned)((ga >> 32) & 0x1ffffffull)) | (2u << 30)));
    v8i g1;
    g1[0] = (int)(2u << 16);   // data_size = 4 bytes
    g1[1] = __builtin_amdgcn_readfirstlane((int)((td0 & 0xffffu) << 16));
    g1[2] = __builtin_amdgcn_readfirstlane((int)((td0 >> 16) | ((td1 & 0xffffu) << 16)));
    g1[3] = __builtin_amdgcn_readfirstlane((int)((td1 >> 16) | (tile0 << 16)));
    g1[4] = 16;                // tile_dim1 = 16 rows
    g1[5] = CO;                // tensor_dim0_stride (low 32)
    g1[6] = 0;
    g1[7] = 0;
    v4i gz; gz[0] = 0; gz[1] = 0; gz[2] = 0; gz[3] = 0;
    v8i gz8; gz8[0] = 0; gz8[1] = 0; gz8[2] = 0; gz8[3] = 0;
    gz8[4] = 0; gz8[5] = 0; gz8[6] = 0; gz8[7] = 0;
    __builtin_amdgcn_tensor_load_to_lds(g0, g1, gz, gz, gz8, 0);
    __builtin_amdgcn_s_wait_tensorcnt(0);
#else
    // fallback: direct guarded loads (branchless clamp+select)
    int lane = threadIdx.x & 31;
    for (int e = lane; e < 512; e += 32) {
        int r = e >> 5, k = e & 31;
        int kc = ((unsigned)k < td0) ? k : 0;
        int rc = ((unsigned)r < td1) ? r : 0;
        int v = gsrc[(size_t)rc * CO + kc];
        ldst[e] = ((unsigned)k < td0 && (unsigned)r < td1 && tile0) ? v : 0;
    }
#endif
}

// ---------------- batched WMMA: out[b] = adj[b](OxO, int01) @ X[b](Oxd, stride 8) -------------
// A tiles streamed via the Tensor Data Mover into LDS (TENSORcnt), f16 WMMA accumulate in f32.
// X must have its padding columns [d,8) zeroed (see zero_pad_cols).
__global__ __launch_bounds__(128) void adj_gemm(const int* __restrict__ adj,
                                                const float* __restrict__ X,
                                                float* __restrict__ outp, int d) {
    constexpr int WPB = 4;
    __shared__ int sAdj[WPB * 16 * 32];      // per-wave 16x32 int tile (TDM destination)
    __shared__ _Float16 sB[32 * 16];
    int b = blockIdx.y;
    int warp = threadIdx.x >> 5, lane = threadIdx.x & 31;
    int m0 = (blockIdx.x * WPB + warp) * 16;
    const int* adjb = adj + (size_t)b * CO * CO;
    const float* xb = X + (size_t)b * CO * 8;
    bool mok = (m0 < CO);
    const int* arow0 = adjb + (size_t)(mok ? m0 : 0) * CO;
    v8f c = {0.f, 0.f, 0.f, 0.f, 0.f, 0.f, 0.f, 0.f};
    int hb = (lane >> 4) * 8;
    int lkn = lane & 15;
    int lr = lane & 15;
    int* myA = &sAdj[warp * 512];
    unsigned td1 = (unsigned)(mok ? (CO - m0) : 1);
    unsigned tile0 = mok ? 32u : 0u;         // tile_dim0==0 => NOP for idle waves

    for (int kb = 0; kb < CO; kb += 32) {
        // stage B tile (32 x 16): branch is wave-uniform; main path is unguarded
        if (kb + 32 <= CO) {
            for (int idx = threadIdx.x; idx < 512; idx += blockDim.x) {
                int kk = idx >> 4, nn = idx & 15;
                float v = (nn < 8) ? xb[(size_t)(kb + kk) * 8 + (nn & 7)] : 0.f;
                sB[idx] = (_Float16)v;
            }
        } else {
            for (int idx = threadIdx.x; idx < 512; idx += blockDim.x) {
                int kk = idx >> 4, nn = idx & 15;
                int kg = kb + kk;
                int kgc = kg < CO ? kg : CO - 1;
                float v = xb[(size_t)kgc * 8 + (nn & 7)];
                v = (kg < CO && nn < 8) ? v : 0.f;
                sB[idx] = (_Float16)v;
            }
        }
        __syncthreads();

        // Tensor Data Mover: pull this wave's 16x32 adjacency tile into LDS
        tdm_load_tile(arow0 + kb, myA, (unsigned)(CO - kb), td1, tile0);

        v16h a, bm;
#pragma unroll
        for (int u = 0; u < 4; ++u) {
            int k0 = hb + 2 * u;
            int k1 = 16 + hb + 2 * u;
            a[2 * u]         = (_Float16)(float)myA[lr * 32 + k0];
            a[2 * u + 1]     = (_Float16)(float)myA[lr * 32 + k0 + 1];
            a[8 + 2 * u]     = (_Float16)(float)myA[lr * 32 + k1];
            a[8 + 2 * u + 1] = (_Float16)(float)myA[lr * 32 + k1 + 1];
            bm[2 * u]         = sB[(k0) * 16 + lkn];
            bm[2 * u + 1]     = sB[(k0 + 1) * 16 + lkn];
            bm[8 + 2 * u]     = sB[(k1) * 16 + lkn];
            bm[8 + 2 * u + 1] = sB[(k1 + 1) * 16 + lkn];
        }
        c = __builtin_amdgcn_wmma_f32_16x16x32_f16(false, a, false, bm, (short)0, c,
                                                   false, false);
        __syncthreads();
    }
    int nc = lane & 15;
    int mbase = (lane >> 4) * 8;
    if (nc < 8) {
        for (int r = 0; r < 8; ++r) {
            int mr = m0 + mbase + r;
            if (mr < CO)
                outp[(size_t)b * CO * 8 + (size_t)mr * 8 + nc] = (nc < d) ? c[r] : 0.f;
        }
    }
}

// ---------------- generic WMMA GEMM: C = act(A @ W + bias), compile-time K,N ----------------
// blockDim = 32*ceil(N/16), grid.x = ceil(Mrows/16). act: 0=none,1=elu,2=tanh
template <int K, int N>
__global__ __launch_bounds__(256) void gemm_wmma(const float* __restrict__ A, int lda,
                                                 const float* __restrict__ W,
                                                 const float* __restrict__ bias,
                                                 float* __restrict__ Cp, int ldc,
                                                 int Mrows, int act) {
    constexpr int KP = (K + 31) & ~31;
    constexpr int NP = (N + 15) & ~15;
    __shared__ _Float16 sA[16 * KP];
    __shared__ _Float16 sW[KP * NP];
    int warp = threadIdx.x >> 5, lane = threadIdx.x & 31;
    int m0 = blockIdx.x * 16;
    int n0 = warp * 16;
    for (int idx = threadIdx.x; idx < 16 * KP; idx += blockDim.x) {
        int r = idx / KP, k = idx - r * KP;
        int mr = m0 + r;
        int mrc = mr < Mrows ? mr : Mrows - 1;
        int kc = k < K ? k : K - 1;
        float v = A[(size_t)mrc * lda + kc];
        v = (mr < Mrows && k < K) ? v : 0.f;
        sA[idx] = (_Float16)v;
    }
    for (int idx = threadIdx.x; idx < KP * NP; idx += blockDim.x) {
        int k = idx / NP, n = idx - k * NP;
        int kc = k < K ? k : K - 1;
        int ncl = n < N ? n : N - 1;
        float v = W[(size_t)kc * N + ncl];
        v = (k < K && n < N) ? v : 0.f;
        sW[idx] = (_Float16)v;
    }
    __syncthreads();
    v8f c = {0.f, 0.f, 0.f, 0.f, 0.f, 0.f, 0.f, 0.f};
    int mloc = lane & 15;
    int hb = (lane >> 4) * 8;
    int nc = n0 + (lane & 15);
#pragma unroll
    for (int kb = 0; kb < KP; kb += 32) {
        v16h a, bm;
#pragma unroll
        for (int u = 0; u < 4; ++u) {
            int k0 = kb + hb + 2 * u;
            int k1 = kb + 16 + hb + 2 * u;
            a[2 * u]         = sA[mloc * KP + k0];
            a[2 * u + 1]     = sA[mloc * KP + k0 + 1];
            a[8 + 2 * u]     = sA[mloc * KP + k1];
            a[8 + 2 * u + 1] = sA[mloc * KP + k1 + 1];
            bm[2 * u]         = sW[k0 * NP + nc];
            bm[2 * u + 1]     = sW[(k0 + 1) * NP + nc];
            bm[8 + 2 * u]     = sW[k1 * NP + nc];
            bm[8 + 2 * u + 1] = sW[(k1 + 1) * NP + nc];
        }
        c = __builtin_amdgcn_wmma_f32_16x16x32_f16(false, a, false, bm, (short)0, c,
                                                   false, false);
    }
    if (nc < N) {
        float bv = bias[nc];
        int mbase = (lane >> 4) * 8;
        for (int r = 0; r < 8; ++r) {
            int mr = m0 + mbase + r;
            if (mr < Mrows) {
                float val = c[r] + bv;
                if (act == 1)      val = val > 0.f ? val : (expf(val) - 1.f);
                else if (act == 2) val = tanhf(val);
                Cp[(size_t)mr * ldc + nc] = val;
            }
        }
    }
}

// ---------------- pooled means ----------------
__global__ void pool_mean(const float* __restrict__ hopes, const float* __restrict__ hmas,
                          float* __restrict__ pools) {
    int b = blockIdx.x;
    __shared__ float red[256];
    for (int j = 0; j < 8; ++j) {
        float s = 0.f;
        for (int i = threadIdx.x; i < CO; i += blockDim.x)
            s += hopes[((size_t)b * CO + i) * 8 + j];
        float t = blk_sum(s, red);
        if (threadIdx.x == 0) pools[b * 8 + j] = t / CO;
        s = 0.f;
        for (int i = threadIdx.x; i < CM; i += blockDim.x)
            s += hmas[((size_t)b * CM + i) * 8 + j];
        t = blk_sum(s, red);
        if (threadIdx.x == 0) pools[CB * 8 + b * 8 + j] = t / CM;
    }
}

// ---------------- build actor input rows (b, m, j) + eligibility ----------------
__global__ void actor_gather(const float* __restrict__ hopes, const float* __restrict__ hmas,
                             const float* __restrict__ pools, const int* __restrict__ ope_step,
                             const int* __restrict__ end_ope, const int* __restrict__ adj_ma,
                             const unsigned char* __restrict__ mjp,
                             const unsigned char* __restrict__ mjf,
                             const unsigned char* __restrict__ mmp,
                             float* __restrict__ X, float* __restrict__ maskf) {
    int flat = blockIdx.x * blockDim.x + threadIdx.x;
    if (flat >= CB * CM * CJ) return;
    int b = flat / (CM * CJ);
    int rem = flat % (CM * CJ);
    int m = rem / CJ, j = rem % CJ;
    int st = ope_step[b * CJ + j], eb = end_ope[b * CJ + j];
    int step = st < eb ? st : eb;
    float* row = X + (size_t)flat * 32;
    for (int k = 0; k < 8; ++k) row[k]      = hopes[((size_t)b * CO + step) * 8 + k];
    for (int k = 0; k < 8; ++k) row[8 + k]  = hmas[((size_t)b * CM + m) * 8 + k];
    for (int k = 0; k < 8; ++k) row[16 + k] = pools[b * 8 + k];
    for (int k = 0; k < 8; ++k) row[24 + k] = pools[CB * 8 + b * 8 + k];
    bool elig = (mjp[b * CJ + j] == 0) && (mjf[b * CJ + j] == 0) && (mmp[b * CM + m] == 0) &&
                (adj_ma[((size_t)b * CO + step) * CM + m] == 1);
    maskf[flat] = elig ? 1.f : 0.f;
}

// ---------------- actor final 64->1 ----------------
__global__ void dot_out(const float* __restrict__ H, const float* __restrict__ w,
                        const float* __restrict__ bias, float* __restrict__ score, int n) {
    int i = blockIdx.x * blockDim.x + threadIdx.x;
    if (i >= n) return;
    float s = bias[0];
    for (int k = 0; k < 64; ++k) s += H[(size_t)i * 64 + k] * w[k];
    score[i] = s;
}

// ---------------- masked softmax per batch ----------------
__global__ void softmax_mask(const float* __restrict__ score, const float* __restrict__ maskf,
                             float* __restrict__ out, int n) {
    int b = blockIdx.x;
    __shared__ float red[256];
    float lmax = -1e30f;
    for (int i = threadIdx.x; i < n; i += blockDim.x)
        if (maskf[(size_t)b * n + i] > 0.5f) lmax = fmaxf(lmax, score[(size_t)b * n + i]);
    float gmax = blk_max(lmax, red);
    float lsum = 0.f;
    for (int i = threadIdx.x; i < n; i += blockDim.x)
        if (maskf[(size_t)b * n + i] > 0.5f) lsum += expf(score[(size_t)b * n + i] - gmax);
    float gsum = blk_sum(lsum, red);
    for (int i = threadIdx.x; i < n; i += blockDim.x) {
        bool e = maskf[(size_t)b * n + i] > 0.5f;
        out[(size_t)b * n + i] = (e && gsum > 0.f) ? expf(score[(size_t)b * n + i] - gmax) / gsum
                                                   : 0.f;
    }
}

extern "C" void kernel_launch(void* const* d_in, const int* in_sizes, int n_in,
                              void* d_out, int out_size, void* d_ws, size_t ws_size,
                              hipStream_t stream) {
    const float* feat_opes = (const float*)d_in[0];
    const float* feat_mas  = (const float*)d_in[1];
    const float* proc      = (const float*)d_in[2];
    const int* adj_ma      = (const int*)d_in[3];
    const int* adj_pre     = (const int*)d_in[4];
    const int* adj_sub     = (const int*)d_in[5];
    const int* ope_step    = (const int*)d_in[6];
    const int* end_ope     = (const int*)d_in[7];
    const unsigned char* m_ma = (const unsigned char*)d_in[8];
    const unsigned char* m_jp = (const unsigned char*)d_in[9];
    const unsigned char* m_jf = (const unsigned char*)d_in[10];
    auto P = [&](int i) { return (const float*)d_in[i]; };
    // params leaf order (jax pytree; dict keys sorted): actor, gat[2], mlps[2]
    // actor: 11..16 (l1W,l1b,l2W,l2b,l3W,l3b)
    // gat layer L: base 17+6L: Wdst, Wedge, Wsrc, ae, al, ar
    // mlps layer L: base 29+30L: proj(l1W..l3b) then sims[0..3](l1W..l3b)

    float* ws = (float*)d_ws;
    size_t off = 0;
    auto alloc = [&](size_t n) { float* p = ws + off; off += n; return p; };
    const int BO = CB * CO;
    float* X_OPE = alloc((size_t)BO * 8);
    float* X_MA  = alloc((size_t)CB * CM * 8);
    float* FS    = alloc((size_t)BO * 8);
    float* EL    = alloc((size_t)BO);
    float* HMAS  = alloc((size_t)CB * CM * 8);
    float* AGG0  = alloc((size_t)BO * 8);
    float* AGG1  = alloc((size_t)BO * 8);
    float* AGG2  = alloc((size_t)BO * 8);
    float* HID1  = alloc((size_t)BO * 128);
    float* HID2  = alloc((size_t)BO * 128);
    float* CAT   = alloc((size_t)BO * 32);
    float* HOPES = alloc((size_t)BO * 8);
    float* POOLS = alloc((size_t)2 * CB * 8);
    const int NA = CB * CM * CJ;
    float* ACTX  = alloc((size_t)NA * 32);
    float* AH1   = alloc((size_t)NA * 64);
    float* AH2   = alloc((size_t)NA * 64);
    float* SCORE = alloc((size_t)NA);
    float* MASKF = alloc((size_t)NA);
    float* STATS = alloc(4);

    norm_nodes<<<CB * 6, 256, 0, stream>>>(feat_opes, X_OPE, CO, 6);
    norm_nodes<<<CB * 3, 256, 0, stream>>>(feat_mas, X_MA, CM, 3);
    zero_pad_cols<<<(BO + 255) / 256, 256, 0, stream>>>(X_OPE, BO, 6);
    zero_pad_cols<<<(CB * CM + 255) / 256, 256, 0, stream>>>(X_MA, CB * CM, 3);
    proc_stats<<<1, 256, 0, stream>>>(proc, STATS, CB * CO * CM);

    for (int lyr = 0; lyr < 2; ++lyr) {
        int din_o = (lyr == 0) ? 6 : 8;
        int din_m = (lyr == 0) ? 3 : 8;
        const float* xo = (lyr == 0) ? X_OPE : HOPES;
        const float* xm = (lyr == 0) ? X_MA : HMAS;
        int g = 17 + 6 * lyr;
        gat_src<<<(BO + 255) / 256, 256, 0, stream>>>(xo, P(g + 2), P(g + 4), FS, EL, BO, din_o);
        gat_attn<<<CB * CM, 256, 0, stream>>>(xm, P(g + 0), P(g + 1), P(g + 3), P(g + 5),
                                              FS, EL, proc, adj_ma, STATS, HMAS, din_m);
        agg_ma<<<(BO + 255) / 256, 256, 0, stream>>>(adj_ma, HMAS, AGG0);
        adj_gemm<<<dim3(32, CB), 128, 0, stream>>>(adj_pre, xo, AGG1, din_o);
        adj_gemm<<<dim3(32, CB), 128, 0, stream>>>(adj_sub, xo, AGG2, din_o);

        int mp = 29 + 30 * lyr;
        const float* aggs[4] = {AGG0, AGG1, AGG2, xo};
        for (int i = 0; i < 4; ++i) {
            int sb = mp + 6 + 6 * i;
            if (i == 0 || lyr == 1) {
                gemm_wmma<8, 128><<<BO / 16, 256, 0, stream>>>(aggs[i], 8, P(sb), P(sb + 1),
                                                               HID1, 128, BO, 1);
            } else {
                gemm_wmma<6, 128><<<BO / 16, 256, 0, stream>>>(aggs[i], 8, P(sb), P(sb + 1),
                                                               HID1, 128, BO, 1);
            }
            gemm_wmma<128, 128><<<BO / 16, 256, 0, stream>>>(HID1, 128, P(sb + 2), P(sb + 3),
                                                             HID2, 128, BO, 1);
            // elu(concat(embs)) == concat(elu(emb_i)): fuse elu + strided concat write
            gemm_wmma<128, 8><<<BO / 16, 32, 0, stream>>>(HID2, 128, P(sb + 4), P(sb + 5),
                                                          CAT + 8 * i, 32, BO, 1);
        }
        gemm_wmma<32, 128><<<BO / 16, 256, 0, stream>>>(CAT, 32, P(mp), P(mp + 1),
                                                        HID1, 128, BO, 1);
        gemm_wmma<128, 128><<<BO / 16, 256, 0, stream>>>(HID1, 128, P(mp + 2), P(mp + 3),
                                                         HID2, 128, BO, 1);
        gemm_wmma<128, 8><<<BO / 16, 32, 0, stream>>>(HID2, 128, P(mp + 4), P(mp + 5),
                                                      HOPES, 8, BO, 0);
    }

    pool_mean<<<CB, 256, 0, stream>>>(HOPES, HMAS, POOLS);
    actor_gather<<<(NA + 255) / 256, 256, 0, stream>>>(HOPES, HMAS, POOLS, ope_step, end_ope,
                                                       adj_ma, m_jp, m_jf, m_ma, ACTX, MASKF);
    gemm_wmma<32, 64><<<NA / 16, 128, 0, stream>>>(ACTX, 32, P(11), P(12), AH1, 64, NA, 2);
    gemm_wmma<64, 64><<<NA / 16, 128, 0, stream>>>(AH1, 64, P(13), P(14), AH2, 64, NA, 2);
    dot_out<<<(NA + 255) / 256, 256, 0, stream>>>(AH2, P(15), P(16), SCORE, NA);
    softmax_mask<<<CB, 256, 0, stream>>>(SCORE, MASKF, (float*)d_out, CM * CJ);
}
